// KANLayer_38208029065329
// MI455X (gfx1250) — matrix operand order repaired and checked by reading, not compiled, not run on hardware
//
#include <hip/hip_runtime.h>

// CDNA5 WMMA vector types
typedef __attribute__((ext_vector_type(16))) __bf16 v16bf;
typedef __attribute__((ext_vector_type(8)))  float  v8f;

namespace {

constexpr int BATCH    = 131072;
constexpr int IN_F     = 128;
constexpr int OUT_F    = 32;
constexpr int NBASIS   = 8;                       // GRID_SIZE + SPLINE_ORDER
constexpr int K_TOT    = IN_F + IN_F * NBASIS;    // 128 silu + 1024 spline = 1152
constexpr int NPAIRS   = K_TOT / 2;               // 576 bf16-pairs per output column
constexpr int WSTRIDE  = 580;                     // padded pair-stride (580%64=4 -> conflict-free, 16B aligned)
constexpr int XSTRIDE  = IN_F + 4;                // padded x row stride (dwords)
constexpr int ROWS_BLK = 128;                     // batch rows per block (8 waves x 16)
constexpr int THREADS  = 256;

constexpr int LDS_X_B   = ROWS_BLK * XSTRIDE * 4;        // 67584 B
constexpr int LDS_W_B   = OUT_F * WSTRIDE * 4;           // 74240 B
constexpr int LDS_BYTES = LDS_X_B + LDS_W_B + OUT_F * 4; // 141952 B < 320 KB WGP LDS

constexpr float LOG2E   = 1.4426950408889634f;
constexpr float SPACING = 2.0f / 5.0f + 1e-6f;
// basis(x) = exp(-0.5*((x-c)/h)^2) = exp2(-((x*S - c*S)^2)), S = sqrt(0.5*log2e)/h
constexpr float BSCALE  = 0.84932196f / SPACING;                 // sqrt(0.5*log2e) / spacing
constexpr float CS0     = (2.0f * 3.0f / 11.0f - 1.0f) * BSCALE; // scaled first center
constexpr float DELTA   = (2.0f / 11.0f) * BSCALE;               // scaled center spacing
constexpr float TWOD    = 2.0f * DELTA;

__device__ __forceinline__ unsigned pack2bf(float lo, float hi) {
  unsigned short l = __builtin_bit_cast(unsigned short, (__bf16)lo);
  unsigned short h = __builtin_bit_cast(unsigned short, (__bf16)hi);
  return ((unsigned)h << 16) | (unsigned)l;
}

struct BFrag { uint4 lo, hi; };  // 32 bytes = 16 bf16 = one B fragment's lane data

__device__ __forceinline__ v16bf load_bfrag(const unsigned* wc, int pairbase) {
  BFrag f;
  f.lo = *(const uint4*)(wc + pairbase);      // ds_load_b128
  f.hi = *(const uint4*)(wc + pairbase + 8);  // ds_load_b128
  return __builtin_bit_cast(v16bf, f);
}

// Fast silu.  gfx1250 has V_TANH_F32: silu(x) = h + h*tanh(h), h = x/2
// (1 mul + 1 trans + 1 fma).  Fallback: x * rcp(1 + exp2(-x*log2e)).
__device__ __forceinline__ float silu_f(float v) {
#if __has_builtin(__builtin_amdgcn_tanhf)
  float h = 0.5f * v;
  return __builtin_fmaf(h, __builtin_amdgcn_tanhf(h), h);
#else
  float e = __builtin_amdgcn_exp2f(-v * LOG2E);
  return v * __builtin_amdgcn_rcpf(1.0f + e);
#endif
}

} // namespace

// Fused KAN layer:  out[b,j] = silu(x)@(ew*bw) + sum_i(ew*bb) + basis(x)@(ew*coeff)
// One bf16 WMMA GEMM, M=BATCH, N=32, K=1152; A generated on the fly from x.
// Gaussian basis factorization: basis_k = G0 * E^k * C_k; the constant C_k is
// folded into the staged B matrix, so the A-side is just a multiply chain.
extern "C" __global__ void __launch_bounds__(THREADS, 1)
kan_wmma_kernel(const float* __restrict__ x,
                const float* __restrict__ coeff,      // [IN][OUT][NBASIS]
                const float* __restrict__ base_w,     // [IN][OUT]
                const float* __restrict__ base_b,     // [IN][OUT]
                const float* __restrict__ edge_w,     // [IN][OUT]
                float* __restrict__ out)              // [BATCH][OUT]
{
  extern __shared__ char smem[];
  float*    lds_x    = (float*)smem;                       // [128][XSTRIDE] fp32
  unsigned* lds_w    = (unsigned*)(smem + LDS_X_B);        // [32 cols][WSTRIDE] packed bf16 pairs
  float*    lds_bias = (float*)(smem + LDS_X_B + LDS_W_B); // [32]

  const int    t    = threadIdx.x;
  const size_t row0 = (size_t)blockIdx.x * ROWS_BLK;

  // ---- stage x tile: 128 rows x 128 feats, fully coalesced float4 ----
  {
    const float* xg = x + row0 * IN_F;
    #pragma unroll
    for (int it = 0; it < (ROWS_BLK * IN_F) / (THREADS * 4); ++it) {
      const int g = (it * THREADS + t) * 4;
      float4 v = *(const float4*)(xg + g);
      const int r = g >> 7;            // /128
      const int f = g & (IN_F - 1);
      *(float4*)(lds_x + r * XSTRIDE + f) = v;
    }
  }

  // C_k = 2^{-(k*DELTA)^2}: folded into the spline weights below
  float ck[NBASIS];
  #pragma unroll
  for (int k = 0; k < NBASIS; ++k) {
    const float kd = (float)k * DELTA;
    ck[k] = __builtin_amdgcn_exp2f(-(kd * kd));
  }

  // ---- stage fused weight matrix as packed bf16 pairs, per-column layout ----
  // K ordering: K<128 -> silu feature K ; K>=128 -> spline (i = (K-128)/8, k = (K-128)%8)
  for (int e = t; e < OUT_F * NPAIRS; e += THREADS) {
    const int col = e / NPAIRS;
    const int p   = e - col * NPAIRS;
    const int k0  = 2 * p;
    float w0, w1;
    if (k0 < IN_F) {
      w0 = edge_w[k0 * OUT_F + col]       * base_w[k0 * OUT_F + col];
      w1 = edge_w[(k0 + 1) * OUT_F + col] * base_w[(k0 + 1) * OUT_F + col];
    } else {
      const int kk = k0 - IN_F;          // even -> same i for both halves
      const int i  = kk >> 3;
      const int b  = kk & 7;
      const float ew = edge_w[i * OUT_F + col];
      w0 = ew * ck[b]     * coeff[(i * OUT_F + col) * NBASIS + b];
      w1 = ew * ck[b + 1] * coeff[(i * OUT_F + col) * NBASIS + b + 1];
    }
    lds_w[col * WSTRIDE + p] = pack2bf(w0, w1);
  }

  // ---- bias column (folded into accumulator init) ----
  if (t < OUT_F) {
    float s = 0.f;
    for (int i = 0; i < IN_F; ++i)
      s += edge_w[i * OUT_F + t] * base_b[i * OUT_F + t];
    lds_bias[t] = s;
  }
  __syncthreads();

  // ---- per-wave GEMM: 16 rows x 32 cols, K = 1152 ----
  const int lane = t & 31;
  const int wid  = t >> 5;
  const int lrow = lane & 15;   // A row index (M = lrow), B/C column within tile
  const int sel  = lane >> 4;   // lower/upper K half of the lane group

  const float*    xrow = lds_x + (wid * 16 + lrow) * XSTRIDE;
  const unsigned* wc0  = lds_w + lrow * WSTRIDE;          // N tile 0: cols 0..15
  const unsigned* wc1  = lds_w + (16 + lrow) * WSTRIDE;   // N tile 1: cols 16..31

  v8f acc0, acc1;
  {
    const float b0 = lds_bias[lrow];
    const float b1 = lds_bias[16 + lrow];
    #pragma unroll
    for (int r = 0; r < 8; ++r) { acc0[r] = b0; acc1[r] = b1; }
  }

  // ---- silu part: K chunks 0..3 (features 32c .. 32c+31) ----
  #pragma unroll
  for (int c = 0; c < 4; ++c) {
    const int f0 = 32 * c + 8 * sel;   // lanes 0-15 -> K 0..7,16..23, lanes 16-31 -> K 8..15,24..31
    const float4 p0 = *(const float4*)(xrow + f0);
    const float4 p1 = *(const float4*)(xrow + f0 + 4);
    const float4 p2 = *(const float4*)(xrow + f0 + 16);
    const float4 p3 = *(const float4*)(xrow + f0 + 20);
    v16bf a;
    a[0]  = (__bf16)silu_f(p0.x); a[1]  = (__bf16)silu_f(p0.y);
    a[2]  = (__bf16)silu_f(p0.z); a[3]  = (__bf16)silu_f(p0.w);
    a[4]  = (__bf16)silu_f(p1.x); a[5]  = (__bf16)silu_f(p1.y);
    a[6]  = (__bf16)silu_f(p1.z); a[7]  = (__bf16)silu_f(p1.w);
    a[8]  = (__bf16)silu_f(p2.x); a[9]  = (__bf16)silu_f(p2.y);
    a[10] = (__bf16)silu_f(p2.z); a[11] = (__bf16)silu_f(p2.w);
    a[12] = (__bf16)silu_f(p3.x); a[13] = (__bf16)silu_f(p3.y);
    a[14] = (__bf16)silu_f(p3.z); a[15] = (__bf16)silu_f(p3.w);

    const int pb = 16 * c + 4 * sel;
    v16bf b0 = load_bfrag(wc0, pb);
    v16bf b1 = load_bfrag(wc1, pb);
    acc0 = __builtin_amdgcn_wmma_f32_16x16x32_bf16(false, a, false, b0, (short)0, acc0, false, false);
    acc1 = __builtin_amdgcn_wmma_f32_16x16x32_bf16(false, a, false, b1, (short)0, acc1, false, false);
  }

  // ---- spline part: K chunks 0..31 (K = 128 + i*8 + k, 4 features per chunk) ----
  // A value = G0 * E^k with G0 = 2^{-u0^2}, E = 2^{2*delta*u0}, u0 = x*BSCALE - CS0.
  // (C_k is already inside the B matrix.)  2 v_exp_f32 + 7 pk-muls per feature pair.
  #pragma unroll 8
  for (int sc = 0; sc < 32; ++sc) {
    const float u0 = __builtin_fmaf(xrow[4 * sc + sel],     BSCALE, -CS0); // halves 0..7
    const float u1 = __builtin_fmaf(xrow[4 * sc + sel + 2], BSCALE, -CS0); // halves 8..15
    float p0 = __builtin_amdgcn_exp2f(-(u0 * u0));   // G0
    float p1 = __builtin_amdgcn_exp2f(-(u1 * u1));
    const float e0 = __builtin_amdgcn_exp2f(u0 * TWOD);  // E
    const float e1 = __builtin_amdgcn_exp2f(u1 * TWOD);
    v16bf a;
    a[0] = (__bf16)p0;
    a[8] = (__bf16)p1;
    #pragma unroll
    for (int k = 1; k < NBASIS; ++k) {
      p0 *= e0;                       // E^k chain (packs as v_pk_mul_f32)
      p1 *= e1;
      a[k]     = (__bf16)p0;
      a[8 + k] = (__bf16)p1;
    }
    const int pb = 64 + 16 * sc + 4 * sel;
    v16bf b0 = load_bfrag(wc0, pb);
    v16bf b1 = load_bfrag(wc1, pb);
    acc0 = __builtin_amdgcn_wmma_f32_16x16x32_bf16(false, a, false, b0, (short)0, acc0, false, false);
    acc1 = __builtin_amdgcn_wmma_f32_16x16x32_bf16(false, a, false, b1, (short)0, acc1, false, false);
  }

  // ---- store C/D (16x16 f32 layout: VGPR r -> M=r (lanes 0-15) / M=r+8 (lanes 16-31)) ----
  float* og = out + (row0 + (size_t)wid * 16) * OUT_F;
  #pragma unroll
  for (int r = 0; r < 8; ++r) {
    const int mo = r + 8 * sel;
    og[mo * OUT_F + lrow]      = acc0[r];
    og[mo * OUT_F + 16 + lrow] = acc1[r];
  }
}

extern "C" void kernel_launch(void* const* d_in, const int* in_sizes, int n_in,
                              void* d_out, int out_size, void* d_ws, size_t ws_size,
                              hipStream_t stream) {
  (void)in_sizes; (void)n_in; (void)d_ws; (void)ws_size; (void)out_size;
  const float* x      = (const float*)d_in[0];
  const float* coeff  = (const float*)d_in[1];
  const float* base_w = (const float*)d_in[2];
  const float* base_b = (const float*)d_in[3];
  const float* edge_w = (const float*)d_in[4];
  float* out = (float*)d_out;

  (void)hipFuncSetAttribute((const void*)kan_wmma_kernel,
                            hipFuncAttributeMaxDynamicSharedMemorySize, LDS_BYTES);

  dim3 grid(BATCH / ROWS_BLK);
  kan_wmma_kernel<<<grid, THREADS, LDS_BYTES, stream>>>(x, coeff, base_w, base_b, edge_w, out);
}
